// DilationConvLayer_12232066859260
// MI455X (gfx1250) — matrix operand order repaired and checked by reading, not compiled, not run on hardware
//
#include <hip/hip_runtime.h>

#define NCH 64      // residual channels C
#define NSK 256     // skip channels S
#define NBATCH 16   // B
#define NT 4096     // T
#define TN 128      // time tile per workgroup
#define XS 72       // padded LDS channel stride (bf16 elements, 144B rows)
#define NLAYER 30
#define GUARD 512   // zero guard rows per batch in bf16 mirror (= max dilation)
#define MROWS (NT + GUARD)

// packed-weight fragment pool: fragment = 32 lanes x 16 bf16 (512 elems, 1KB)
// dil : fid = layer*32 + tap*16 + mb*2 + kb          (fids    0.. 959)
// res : fid = 960  + layer*8  + mb*2 + kb            (fids  960..1199)
// skip: fid = 1200 + layer*32 + mb*2 + kb            (fids 1200..2159)
#define NFRAG 2160

typedef __attribute__((ext_vector_type(16))) __bf16 v16bf;
typedef __attribute__((ext_vector_type(8)))  __bf16 v8bf;
typedef __attribute__((ext_vector_type(2)))  __bf16 v2bf;
typedef __attribute__((ext_vector_type(8)))  float  v8f;

static __device__ __forceinline__ v8f wmma_bf16(v16bf a, v16bf b, v8f c) {
  // D(f32 16x16) = A(bf16 16x32) * B(bf16 32x16) + C
  return __builtin_amdgcn_wmma_f32_16x16x32_bf16(false, a, false, b, (short)0, c,
                                                 false, false);
}

// CDNA5 has a hardware v_tanh_f32 transcendental; use it when exposed.
static __device__ __forceinline__ float fast_tanh(float x) {
#if __has_builtin(__builtin_amdgcn_tanhf)
  return __builtin_amdgcn_tanhf(x);
#elif __has_builtin(__builtin_amdgcn_tanh_f32)
  return __builtin_amdgcn_tanh_f32(x);
#else
  return tanhf(x);
#endif
}
// sigmoid(x) == 0.5*tanh(x/2) + 0.5 (exact identity) -> 1 trans + 1 fma
static __device__ __forceinline__ float fast_sigmoid(float x) {
#if __has_builtin(__builtin_amdgcn_tanhf) || __has_builtin(__builtin_amdgcn_tanh_f32)
  return fmaf(0.5f, fast_tanh(0.5f * x), 0.5f);
#else
  return 1.0f / (1.0f + __expf(-x));
#endif
}

// A fragment load from the pre-swizzled bf16 pool: one 32B contiguous read/lane.
static __device__ __forceinline__ v16bf load_a_packed(const __bf16* __restrict__ wP,
                                                      int fid) {
  return *(const v16bf*)(wP + (long)fid * 512 + (threadIdx.x & 31) * 16);
}

// B fragment (32K x 16N) from a [t][c] bf16 base with int element offset.
// Contiguous k order used here (A packed to match): lanes 0-15: N=lane, K=0..15; lanes 16-31: N=lane-16, K=16..31.
static __device__ __forceinline__ v16bf load_b_off(const __bf16* __restrict__ base,
                                                   int off, int kb) {
  const int koff = kb * 32 + (((threadIdx.x & 31) >> 4) ? 16 : 0);
  const __bf16* p = base + off + koff;       // 16B-aligned
  v8bf lo = *(const v8bf*)p;
  v8bf hi = *(const v8bf*)(p + 8);
  return __builtin_shufflevector(lo, hi, 0,1,2,3,4,5,6,7,8,9,10,11,12,13,14,15);
}

// B fragment from LDS z-tile laid out [t][c], stride XS.
static __device__ __forceinline__ v16bf load_b_lds(const __bf16* __restrict__ lds,
                                                   int n0, int kb) {
  const int lane = threadIdx.x & 31;
  const int n    = n0 + (lane & 15);
  const int koff = kb * 32 + ((lane >> 4) ? 16 : 0);
  const __bf16* p = lds + n * XS + koff;     // 16B-aligned
  v8bf lo = *(const v8bf*)p;
  v8bf hi = *(const v8bf*)(p + 8);
  return __builtin_shufflevector(lo, hi, 0,1,2,3,4,5,6,7,8,9,10,11,12,13,14,15);
}

// ---- one-shot weight pre-swizzle into WMMA A-fragment layout ----
// A layout (matches B): lanes 0-15 -> K {k0..k0+15},
//                       lanes 16-31 -> K {k0+16..k0+31}.
__global__ __launch_bounds__(256) void pack_weights(
    const float* __restrict__ dil_w, const float* __restrict__ res_w,
    const float* __restrict__ skip_w, __bf16* __restrict__ wP) {
  const int gid = blockIdx.x * 256 + threadIdx.x;
  if (gid >= NFRAG * 32) return;
  const int fid  = gid >> 5;
  const int lane = gid & 31;

  const float* base;
  int rs, ks, mb, kb;
  if (fid < 960) {                       // dil_w [L][2C][C][2]
    int layer = fid >> 5, r = fid & 31;
    int tap = r >> 4;
    mb = (r >> 1) & 7;  kb = r & 1;
    base = dil_w + (long)layer * (2 * NCH) * NCH * 2 + tap;
    rs = NCH * 2;  ks = 2;
  } else if (fid < 1200) {               // res_w [L][C][C]
    int f = fid - 960;
    int layer = f >> 3;
    mb = (f >> 1) & 3;  kb = f & 1;
    base = res_w + (long)layer * NCH * NCH;
    rs = NCH;  ks = 1;
  } else {                               // skip_w [L][S][C]
    int f = fid - 1200;
    int layer = f >> 5;
    mb = (f >> 1) & 15;  kb = f & 1;
    base = skip_w + (long)layer * NSK * NCH;
    rs = NCH;  ks = 1;
  }

  const int m   = mb * 16 + (lane & 15);
  const int hi8 = (lane >> 4) * 16;
  __bf16* dst = wP + (long)fid * 512 + lane * 16;
#pragma unroll
  for (int i = 0; i < 8; ++i)
    dst[i]     = (__bf16)base[m * rs + (kb * 32 + hi8 + i) * ks];
#pragma unroll
  for (int i = 0; i < 8; ++i)
    dst[8 + i] = (__bf16)base[m * rs + (kb * 32 + 8 + hi8 + i) * ks];
}

// ---- zero the causal guard bands of both bf16 mirrors (once per launch) ----
__global__ __launch_bounds__(256) void zero_guards(__bf16* __restrict__ ma,
                                                   __bf16* __restrict__ mb) {
  const int gid = blockIdx.x * 256 + threadIdx.x;
  const int per_b4 = GUARD * NCH / 4;                 // 8192 x 8B per batch
  if (gid >= NBATCH * per_b4) return;
  const int  b   = gid / per_b4;
  const int  r4  = gid % per_b4;
  const long off = (long)b * MROWS * NCH + (long)r4 * 4;
  *(unsigned long long*)(ma + off) = 0ull;
  *(unsigned long long*)(mb + off) = 0ull;
}

// ---- one-shot transpose of forward_input [b][c][t] -> [b][t][c] (f32 + bf16) ----
__global__ __launch_bounds__(256) void init_tc(const float* __restrict__ x,
                                               float* __restrict__ xtc,
                                               __bf16* __restrict__ xbf) {
  const long gid = (long)blockIdx.x * 256 + threadIdx.x;
  if (gid >= (long)NBATCH * (NCH / 2) * NT) return;
  const int  t  = (int)(gid & (NT - 1));        // fastest -> coalesced reads
  const int  cp = (int)((gid >> 12) & 31);
  const long b  = gid >> 17;
  const int  c  = cp * 2;
  float v0 = x[((long)b * NCH + c) * NT + t];
  float v1 = x[((long)b * NCH + c + 1) * NT + t];
  float2 f2; f2.x = v0; f2.y = v1;
  *(float2*)(xtc + ((long)b * NT + t) * NCH + c) = f2;
  v2bf m2; m2[0] = (__bf16)v0; m2[1] = (__bf16)v1;
  *(v2bf*)(xbf + ((long)b * MROWS + GUARD + t) * NCH + c) = m2;
}

__global__ __launch_bounds__(256) void wavenet_layer(
    const float* __restrict__ x_src,   // f32 residual,      [b][t][c]
    float* __restrict__ x_dst,         // f32 residual out,  [b][t][c]
    const __bf16* __restrict__ xbf_src,// bf16 mirror in,    [b][GUARD+t][c]
    __bf16* __restrict__ xbf_dst,      // bf16 mirror out,   [b][GUARD+t][c]
    const __bf16* __restrict__ wP,
    const float* __restrict__ dil_b, const float* __restrict__ res_b,
    const float* __restrict__ skip_b,
    float* __restrict__ out,           // [b][s][t], atomically accumulated
    int layer, int dilation) {
  __shared__ __align__(32) __bf16 zbuf[TN * XS];  // gated z tile, [t][c]

  const int b    = blockIdx.y;
  const int t0   = blockIdx.x * TN;
  const int tid  = threadIdx.x;
  const int wv   = tid >> 5;
  const int lane = tid & 31;
  const int hi   = lane >> 4;   // upper-half lanes -> M += 8 in C fragments
  const int ln   = lane & 15;   // N index within tile

  const v8f vzero = {0.f, 0.f, 0.f, 0.f, 0.f, 0.f, 0.f, 0.f};

  // uniform per-block base pointers -> SADDR + small int VADDR offsets
  const __bf16* bcur = xbf_src + ((long)b * MROWS + GUARD) * NCH;
  const __bf16* bsh  = bcur - dilation * NCH;   // guard band guarantees zeros
  const float*  xsb  = x_src + (long)b * NT * NCH;
  float*        xdb  = x_dst + (long)b * NT * NCH;
  __bf16*       mdb  = xbf_dst + ((long)b * MROWS + GUARD) * NCH;
  float*        outb = out + (long)b * NSK * NT;

  // ---- dilated conv (M=128, K=64 per tap, 2 taps), B direct from bf16 mirror ----
  // wave wv: channel group cg = wv&3 -> tanh M-block cg, sigmoid M-block cg+4,
  //          time half nh = wv>>2 -> N blocks nh*4 .. nh*4+3
  const int cg = wv & 3;
  const int nh = wv >> 2;

  v16bf aT[2][2], aS[2][2];  // [tap][kb]
#pragma unroll
  for (int tap = 0; tap < 2; ++tap)
#pragma unroll
    for (int kb = 0; kb < 2; ++kb) {
      aT[tap][kb] = load_a_packed(wP, layer * 32 + tap * 16 + cg * 2 + kb);
      aS[tap][kb] = load_a_packed(wP, layer * 32 + tap * 16 + (4 + cg) * 2 + kb);
    }

  v8f accT[4], accS[4];
#pragma unroll
  for (int nb = 0; nb < 4; ++nb) { accT[nb] = vzero; accS[nb] = vzero; }

#pragma unroll
  for (int nb = 0; nb < 4; ++nb) {
    const int t   = t0 + nh * 64 + nb * 16 + ln;  // this lane's time row
    const int off = t * NCH;                      // small int offset
    v16bf bc0 = load_b_off(bcur, off, 0);
    v16bf bc1 = load_b_off(bcur, off, 1);
    v16bf bs0 = load_b_off(bsh,  off, 0);         // x[t-d]; guard rows are zero
    v16bf bs1 = load_b_off(bsh,  off, 1);
    accT[nb] = wmma_bf16(aT[0][0], bs0, accT[nb]);   // tap 0: x[t-d]
    accT[nb] = wmma_bf16(aT[0][1], bs1, accT[nb]);
    accT[nb] = wmma_bf16(aT[1][0], bc0, accT[nb]);   // tap 1: x[t]
    accT[nb] = wmma_bf16(aT[1][1], bc1, accT[nb]);
    accS[nb] = wmma_bf16(aS[0][0], bs0, accS[nb]);
    accS[nb] = wmma_bf16(aS[0][1], bs1, accS[nb]);
    accS[nb] = wmma_bf16(aS[1][0], bc0, accS[nb]);
    accS[nb] = wmma_bf16(aS[1][1], bc1, accS[nb]);
  }

  // gating in-register (tanh/sigmoid fragments align element-for-element)
  // z = tanh(yT + bT) * sigmoid(yS + bS), via hardware v_tanh_f32
  const float* db = dil_b + (long)layer * (2 * NCH);
  float bT[8], bS[8];
#pragma unroll
  for (int r = 0; r < 8; ++r) {
    int c = cg * 16 + r + 8 * hi;
    bT[r] = db[c];
    bS[r] = db[NCH + c];
  }
#pragma unroll
  for (int nb = 0; nb < 4; ++nb) {
    const int t = nh * 64 + nb * 16 + ln;
#pragma unroll
    for (int r = 0; r < 8; r += 2) {
      v2bf z2;
#pragma unroll
      for (int j = 0; j < 2; ++j) {
        float g  = fast_tanh(accT[nb][r + j] + bT[r + j]);
        float sg = fast_sigmoid(accS[nb][r + j] + bS[r + j]);
        z2[j] = (__bf16)(g * sg);
      }
      int c = cg * 16 + r + 8 * hi;        // r even -> c even -> 4B aligned
      *(v2bf*)(&zbuf[t * XS + c]) = z2;
    }
  }
  __syncthreads();

  // ---- residual GEMM: x_dst = x_src + res_w @ z + res_b  (+ bf16 mirror) ----
  {
    const float* rb = res_b + (long)layer * NCH;
    const int n0 = wv * 16;  // each wave owns one N block, all 4 M blocks
    const int t  = t0 + n0 + ln;
    v16bf zb0 = load_b_lds(zbuf, n0, 0);
    v16bf zb1 = load_b_lds(zbuf, n0, 1);
#pragma unroll
    for (int mb = 0; mb < 4; ++mb) {
      v16bf a0 = load_a_packed(wP, 960 + layer * 8 + mb * 2 + 0);
      v16bf a1 = load_a_packed(wP, 960 + layer * 8 + mb * 2 + 1);
      v8f acc = vzero;
      acc = wmma_bf16(a0, zb0, acc);
      acc = wmma_bf16(a1, zb1, acc);
#pragma unroll
      for (int r = 0; r < 8; r += 2) {
        const int c = mb * 16 + r + 8 * hi;          // even -> paired
        const int o = t * NCH + c;                   // small int offset
        float2 xr = *(const float2*)(xsb + o);       // f32 residual carry
        float v0 = xr.x + acc[r]     + rb[c];
        float v1 = xr.y + acc[r + 1] + rb[c + 1];
        float2 xo; xo.x = v0; xo.y = v1;
        *(float2*)(xdb + o) = xo;
        v2bf m2; m2[0] = (__bf16)v0; m2[1] = (__bf16)v1;
        *(v2bf*)(mdb + o) = m2;                      // mirror for next layer's B
      }
    }
  }

  // ---- skip GEMM: out += skip_w @ z + skip_b (atomic, d_out pre-zeroed) ----
  {
    const float* sb = skip_b + (long)layer * NSK;
#pragma unroll
    for (int mi = 0; mi < 2; ++mi) {
      const int mb = wv * 2 + mi;  // 16 M blocks / 8 waves
      v16bf a0 = load_a_packed(wP, 1200 + layer * 32 + mb * 2 + 0);
      v16bf a1 = load_a_packed(wP, 1200 + layer * 32 + mb * 2 + 1);
#pragma unroll
      for (int nb = 0; nb < 8; ++nb) {
        const int n0 = nb * 16;
        v16bf zb0 = load_b_lds(zbuf, n0, 0);
        v16bf zb1 = load_b_lds(zbuf, n0, 1);
        v8f acc = vzero;
        acc = wmma_bf16(a0, zb0, acc);
        acc = wmma_bf16(a1, zb1, acc);
#pragma unroll
        for (int r = 0; r < 8; ++r) {
          const int s  = mb * 16 + r + 8 * hi;
          const int gi = s * NT + t0 + n0 + ln;      // small int offset
          unsafeAtomicAdd(outb + gi, acc[r] + sb[s]);// global_atomic_add_f32
        }
      }
    }
  }
}

extern "C" void kernel_launch(void* const* d_in, const int* in_sizes, int n_in,
                              void* d_out, int out_size, void* d_ws, size_t ws_size,
                              hipStream_t stream) {
  const float* x_in   = (const float*)d_in[0];
  const float* dil_w  = (const float*)d_in[1];
  const float* dil_b  = (const float*)d_in[2];
  const float* res_w  = (const float*)d_in[3];
  const float* res_b  = (const float*)d_in[4];
  const float* skip_w = (const float*)d_in[5];
  const float* skip_b = (const float*)d_in[6];
  float* out = (float*)d_out;

  // workspace layout:
  //   xa, xb : f32 residual ping-pong, [b][t][c]          (2 x 16 MB)
  //   ma, mb : bf16 mirrors with guard, [b][GUARD+t][c]   (2 x ~9.4 MB, L2-resident)
  //   wP     : packed bf16 weight fragments               (~2.2 MB, L2-resident)
  const size_t XELEMS = (size_t)NBATCH * NCH * NT;
  const size_t MELEMS = (size_t)NBATCH * NCH * MROWS;
  float*  xa = (float*)d_ws;
  float*  xb = xa + XELEMS;
  __bf16* ma = (__bf16*)(xb + XELEMS);
  __bf16* mb = ma + MELEMS;
  __bf16* wP = mb + MELEMS;

  // one-shot preprocessing (graph-capture safe, deterministic every call)
  hipMemsetAsync(d_out, 0, (size_t)out_size * sizeof(float), stream);
  pack_weights<<<(NFRAG * 32 + 255) / 256, 256, 0, stream>>>(dil_w, res_w, skip_w, wP);
  zero_guards<<<(NBATCH * (GUARD * NCH / 4) + 255) / 256, 256, 0, stream>>>(ma, mb);
  const long tc_threads = (long)NBATCH * (NCH / 2) * NT;
  init_tc<<<(int)((tc_threads + 255) / 256), 256, 0, stream>>>(x_in, xb, mb);

  dim3 grid(NT / TN, NBATCH);  // 32 x 16 workgroups
  dim3 block(256);             // 8 waves (wave32)

  for (int i = 0; i < NLAYER; ++i) {
    const int d = 1 << (i % 10);
    // layer 0 reads the init buffers (xb/mb) and writes xa/ma; then ping-pong
    const float*  xs = (i & 1) ? xa : xb;
    float*        xd = (i & 1) ? xb : xa;
    const __bf16* ms = (i & 1) ? ma : mb;
    __bf16*       md = (i & 1) ? mb : ma;
    wavenet_layer<<<grid, block, 0, stream>>>(xs, xd, ms, md, wP, dil_b, res_b,
                                              skip_b, out, i, d);
  }
}
